// Transformer_29506425324097
// MI455X (gfx1250) — matrix operand order repaired and checked by reading, not compiled, not run on hardware
//
#include <hip/hip_runtime.h>

// ---------------- problem constants (match reference) ----------------
#define BB   2
#define SS   2048
#define DD   1024
#define HH   16
#define LL   4
#define VV   2000
#define DKK  64
#define DFF  4096
#define EPSV 1e-5f
#define MTOK (BB * SS)   // 4096 token rows
#define NEGF -3.0e38f

// ---------------- CDNA5 WMMA types ----------------
typedef __attribute__((ext_vector_type(16))) __bf16 v16bf;
typedef __attribute__((ext_vector_type(8)))  float  v8f;
typedef __attribute__((ext_vector_type(4)))  unsigned int v4u;
typedef unsigned short u16;

union BfFrag { v16bf v; u16 h[16]; unsigned int u[8]; v4u q[2]; };

__device__ __forceinline__ u16 f2bf(float f) {
  union { float f; unsigned int u; } c; c.f = f;
  unsigned int u = c.u + 0x7FFFu + ((c.u >> 16) & 1u);  // round-to-nearest-even
  return (u16)(u >> 16);
}

// ---------------- DPP cross-lane reductions (no LDS traffic) ----------
#define DPP8SEL(a,b,c,d,e,f,g,h) \
  ((a)|((b)<<3)|((c)<<6)|((d)<<9)|((e)<<12)|((f)<<15)|((g)<<18)|((h)<<21))
#define SEL_X1 DPP8SEL(1,0,3,2,5,4,7,6)
#define SEL_X2 DPP8SEL(2,3,0,1,6,7,4,5)
#define SEL_X4 DPP8SEL(4,5,6,7,0,1,2,3)

template <unsigned S>
__device__ __forceinline__ float dpp8f(float x) {
  return __builtin_bit_cast(float,
      __builtin_amdgcn_mov_dpp8(__builtin_bit_cast(int, x), S));
}
__device__ __forceinline__ float dpp_ror8(float x) {   // lane ^ 8 within row of 16
  return __builtin_bit_cast(float, __builtin_amdgcn_update_dpp(
      0, __builtin_bit_cast(int, x), 0x128 /*row_ror:8*/, 0xf, 0xf, true));
}
__device__ __forceinline__ float redmax16(float m) {   // max over 16-lane group
  m = fmaxf(m, dpp8f<SEL_X1>(m));
  m = fmaxf(m, dpp8f<SEL_X2>(m));
  m = fmaxf(m, dpp8f<SEL_X4>(m));
  m = fmaxf(m, dpp_ror8(m));
  return m;
}
__device__ __forceinline__ float redsum16(float s) {   // sum over 16-lane group
  s += dpp8f<SEL_X1>(s);
  s += dpp8f<SEL_X2>(s);
  s += dpp8f<SEL_X4>(s);
  s += dpp_ror8(s);
  return s;
}

// async global->LDS b128 copy (ASYNCcnt-tracked) -----------------------
__device__ __forceinline__ void async_cp_b128(const u16* g, const u16* lds) {
  unsigned ldsoff = (unsigned)(size_t)lds;              // LDS addr = low 32 bits
  asm volatile("global_load_async_to_lds_b128 %0, %1, off"
               :: "v"(ldsoff), "v"(g) : "memory");
}

// =====================================================================
// Embedding + positional encoding -> x (f32 residual) and xb (bf16)
// =====================================================================
__global__ __launch_bounds__(256)
void embed_pe_kernel(const int* __restrict__ idx, const float* __restrict__ emb,
                     float* __restrict__ x, u16* __restrict__ xb) {
  size_t t = (size_t)blockIdx.x * 256 + threadIdx.x;
  int d = (int)(t & (DD - 1));
  size_t bs = t / DD;
  int s = (int)(bs & (SS - 1));
  int token = idx[bs];
  float fi2 = (float)(d & ~1);
  float angle = (float)s * __expf(fi2 * (-9.210340371976184f / (float)DD));
  float pe = (d & 1) ? __cosf(angle) : __sinf(angle);
  float v = emb[(size_t)token * DD + d] + pe;
  x[t] = v;
  xb[t] = f2bf(v);
}

// =====================================================================
// Weight transpose + convert:  in[N,K] f32  ->  out[K,Npad] bf16 (pad=0)
// =====================================================================
__global__ __launch_bounds__(256)
void wtrans_kernel(const float* __restrict__ in, u16* __restrict__ out,
                   int N, int K, int Npad) {
  __shared__ u16 t[32][33];
  const int k0 = blockIdx.x * 32, n0 = blockIdx.y * 32;
  const int tx = threadIdx.x & 31, ty = threadIdx.x >> 5;
#pragma unroll
  for (int i = 0; i < 4; ++i) {
    int n = n0 + ty + i * 8;
    int nc = n < N ? n : N - 1;
    float v = in[(size_t)nc * K + k0 + tx];
    t[ty + i * 8][tx] = (n < N) ? f2bf(v) : (u16)0;
  }
  __syncthreads();
#pragma unroll
  for (int i = 0; i < 4; ++i)
    out[(size_t)(k0 + ty + i * 8) * Npad + n0 + tx] = t[tx][ty + i * 8];
}

// =====================================================================
// K transpose for attention:  kb [B,S,H,DK] bf16 -> kt [B,H,DK,S] bf16
// =====================================================================
__global__ __launch_bounds__(256)
void kt_trans_kernel(const u16* __restrict__ in, u16* __restrict__ out) {
  __shared__ u16 t[32][33];
  const int s0 = blockIdx.x * 32, d0 = blockIdx.y * 32;
  const int bh = blockIdx.z;
  const int b = bh >> 4, h = bh & 15;
  const u16* ip = in + (size_t)b * SS * (HH * DKK) + (size_t)h * DKK;
  u16* op = out + (size_t)bh * DKK * SS;
  const int tx = threadIdx.x & 31, ty = threadIdx.x >> 5;
#pragma unroll
  for (int i = 0; i < 4; ++i)
    t[ty + i * 8][tx] = ip[(size_t)(s0 + ty + i * 8) * (HH * DKK) + d0 + tx];
  __syncthreads();
#pragma unroll
  for (int i = 0; i < 4; ++i)
    op[(size_t)(d0 + ty + i * 8) * SS + s0 + tx] = t[tx][ty + i * 8];
}

// =====================================================================
// WMMA GEMM with double-buffered async-to-LDS staging.
// C[M,N] = A[M,K](bf16) @ Bt[K,Nld](bf16), f32 accumulate.
// block 256 = 8 waves; block tile 128(M) x 64(N); wave tile 32x32.
// =====================================================================
template <bool BOUND>
__global__ __launch_bounds__(256)
void wmma_gemm(const u16* __restrict__ A, const u16* __restrict__ Bt,
               float* __restrict__ Cf, u16* __restrict__ Cb,
               int N, int Nld, int K, int relu, float oscale) {
  __shared__ u16 As[2][128 * 32];
  __shared__ u16 Bs[2][32 * 64];
  const int tid = threadIdx.x;
  const int lane = tid & 31;
  const int wave = tid >> 5;
  const int rowg = blockIdx.y * 128;
  const int colg = blockIdx.x * 64;
  const int row0 = (wave >> 1) * 32;   // local M offset
  const int col0 = (wave & 1) * 32;    // local N offset
  const int ml = lane & 15, hf = lane >> 4;

  auto issue_tile = [&](int buf, int kc0) {
#pragma unroll
    for (int i = 0; i < 2; ++i) {                    // A: 512 chunks of 16B
      int c = tid + i * 256;
      int r = c >> 2, off = (c & 3) * 8;
      async_cp_b128(A + (size_t)(rowg + r) * K + kc0 + off,
                    &As[buf][r * 32 + off]);
    }
    {                                                // B: 256 chunks of 16B
      int r = tid >> 3, off = (tid & 7) * 8;
      async_cp_b128(Bt + (size_t)(kc0 + r) * Nld + colg + off,
                    &Bs[buf][r * 64 + off]);
    }
  };

  v8f acc[2][2] = {};
  const int nk = K / 32;
  issue_tile(0, 0);
  for (int kc = 0; kc < nk; ++kc) {
    const int cur = kc & 1;
    if (kc + 1 < nk) {
      issue_tile(cur ^ 1, (kc + 1) * 32);
      asm volatile("s_wait_asynccnt 0x3" ::: "memory");   // cur tile landed
    } else {
      asm volatile("s_wait_asynccnt 0x0" ::: "memory");
    }
    __syncthreads();

    BfFrag a[2], b[2];
#pragma unroll
    for (int rt = 0; rt < 2; ++rt) {
      const u16* ar = &As[cur][(row0 + rt * 16 + ml) * 32 + hf * 8];
      a[rt].q[0] = *(const v4u*)ar;
      a[rt].q[1] = *(const v4u*)(ar + 16);
    }
#pragma unroll
    for (int nt = 0; nt < 2; ++nt) {
      const u16* br = &Bs[cur][col0 + nt * 16 + ml];
#pragma unroll
      for (int s = 0; s < 16; ++s)
        b[nt].h[s] = br[((s & 7) + hf * 8 + (s >> 3) * 16) * 64];
    }
#pragma unroll
    for (int rt = 0; rt < 2; ++rt)
#pragma unroll
      for (int nt = 0; nt < 2; ++nt)
        acc[rt][nt] = __builtin_amdgcn_wmma_f32_16x16x32_bf16(
            false, a[rt].v, false, b[nt].v, (short)0, acc[rt][nt], false, false);
    __syncthreads();
  }

  // epilogue: D layout: lane holds N=lane&15; VGPR j holds M=j+8*half
#pragma unroll
  for (int rt = 0; rt < 2; ++rt)
#pragma unroll
    for (int nt = 0; nt < 2; ++nt)
#pragma unroll
      for (int j = 0; j < 8; ++j) {
        int row = rowg + row0 + rt * 16 + j + 8 * hf;
        int col = colg + col0 + nt * 16 + ml;
        float vv = acc[rt][nt][j] * oscale;
        if (relu) vv = fmaxf(vv, 0.f);
        if (!BOUND || col < N) {
          if (Cf) Cf[(size_t)row * N + col] = vv;
          if (Cb) Cb[(size_t)row * N + col] = f2bf(vv);
        }
      }
}

// =====================================================================
// Flash attention (causal), WMMA, all-bf16. 1/sqrt(dk) pre-folded into q.
// Q,V,O: [B,S,H,DK] bf16.  Kt: [B,H,DK,S] bf16.
// =====================================================================
template <bool MASKED>
__device__ __forceinline__ void attn_step(
    int kb0, int q0, int lane, int ml, int hf,
    const BfFrag (&aq)[2][2], const u16* Ktb, const u16* Vb, u16* ldsPw,
    v8f (&acc)[2][4], float (&mrow)[2][8], float (&lrow)[2][8]) {
  // ---------- S = Q K^T over DK=64 ----------
  v8f sacc[2][2] = {};
#pragma unroll
  for (int ci = 0; ci < 2; ++ci) {
    BfFrag bk[2];
#pragma unroll
    for (int nt = 0; nt < 2; ++nt) {
      const u16* br = Ktb + (size_t)(ci * 32) * SS + kb0 + nt * 16 + ml;
#pragma unroll
      for (int s = 0; s < 16; ++s)
        bk[nt].h[s] = br[(size_t)((s & 7) + hf * 8 + (s >> 3) * 16) * SS];
    }
#pragma unroll
    for (int rt = 0; rt < 2; ++rt)
#pragma unroll
      for (int nt = 0; nt < 2; ++nt)
        sacc[rt][nt] = __builtin_amdgcn_wmma_f32_16x16x32_bf16(
            false, aq[ci][rt].v, false, bk[nt].v, (short)0, sacc[rt][nt], false, false);
  }
  // ---------- online softmax (DPP reductions within 16-lane halves) ----
#pragma unroll
  for (int rt = 0; rt < 2; ++rt) {
#pragma unroll
    for (int j = 0; j < 8; ++j) {
      float s0 = sacc[rt][0][j];
      float s1 = sacc[rt][1][j];
      if (MASKED) {
        int qrow = q0 + rt * 16 + j + 8 * hf;
        s0 = (kb0 + ml      <= qrow) ? s0 : NEGF;
        s1 = (kb0 + 16 + ml <= qrow) ? s1 : NEGF;
      }
      float m = redmax16(fmaxf(s0, s1));
      float newm = fmaxf(mrow[rt][j], m);
      float scl = __expf(mrow[rt][j] - newm);
      float p0 = __expf(s0 - newm);
      float p1 = __expf(s1 - newm);
      float rs = redsum16(p0 + p1);
      lrow[rt][j] = lrow[rt][j] * scl + rs;
      mrow[rt][j] = newm;
#pragma unroll
      for (int ct = 0; ct < 4; ++ct) acc[rt][ct][j] *= scl;
      int r = rt * 16 + j + 8 * hf;
      ldsPw[r * 32 + ml]      = f2bf(p0);
      ldsPw[r * 32 + 16 + ml] = f2bf(p1);
    }
  }
  asm volatile("s_wait_dscnt 0" ::: "memory");       // same-wave LDS RAW fence
  // ---------- O += P V ----------
  BfFrag ap[2];
#pragma unroll
  for (int rt = 0; rt < 2; ++rt) {
    int r = rt * 16 + ml;
    ap[rt].q[0] = *(const v4u*)&ldsPw[r * 32 + hf * 8];
    ap[rt].q[1] = *(const v4u*)&ldsPw[r * 32 + 16 + hf * 8];
  }
#pragma unroll
  for (int ct = 0; ct < 4; ++ct) {
    BfFrag bv;
    const u16* vr = Vb + (size_t)kb0 * (HH * DKK) + ct * 16 + ml;
#pragma unroll
    for (int s = 0; s < 16; ++s)
      bv.h[s] = vr[(size_t)((s & 7) + hf * 8 + (s >> 3) * 16) * (HH * DKK)];
#pragma unroll
    for (int rt = 0; rt < 2; ++rt)
      acc[rt][ct] = __builtin_amdgcn_wmma_f32_16x16x32_bf16(
          false, ap[rt].v, false, bv.v, (short)0, acc[rt][ct], false, false);
  }
}

__global__ __launch_bounds__(256)
void attn_kernel(const u16* __restrict__ Q, const u16* __restrict__ Kt,
                 const u16* __restrict__ V, u16* __restrict__ O) {
  __shared__ u16 ldsP[8][32 * 32];
  const int lane = threadIdx.x & 31;
  const int wave = threadIdx.x >> 5;
  const int ml = lane & 15;
  const int hf = lane >> 4;
  const int bidx = blockIdx.z;
  const int h = blockIdx.y;
  const int q0 = blockIdx.x * 256 + wave * 32;
  const int tstr = HH * DKK;
  const u16* Qb = Q + (size_t)bidx * SS * tstr + (size_t)h * DKK;
  const u16* Vb = V + (size_t)bidx * SS * tstr + (size_t)h * DKK;
  u16*       Ob = O + (size_t)bidx * SS * tstr + (size_t)h * DKK;
  const u16* Ktb = Kt + (size_t)(bidx * HH + h) * DKK * SS;

  BfFrag aq[2][2];                                   // hoisted Q fragments
#pragma unroll
  for (int ci = 0; ci < 2; ++ci)
#pragma unroll
    for (int rt = 0; rt < 2; ++rt) {
      const u16* qr = Qb + (size_t)(q0 + rt * 16 + ml) * tstr + ci * 32 + hf * 8;
      aq[ci][rt].q[0] = *(const v4u*)qr;
      aq[ci][rt].q[1] = *(const v4u*)(qr + 16);
    }

  v8f acc[2][4] = {};
  float mrow[2][8], lrow[2][8];
#pragma unroll
  for (int rt = 0; rt < 2; ++rt)
#pragma unroll
    for (int j = 0; j < 8; ++j) { mrow[rt][j] = NEGF; lrow[rt][j] = 0.f; }

  for (int kb0 = 0; kb0 < q0; kb0 += 32)             // full (unmasked) blocks
    attn_step<false>(kb0, q0, lane, ml, hf, aq, Ktb, Vb, ldsP[wave], acc, mrow, lrow);
  attn_step<true>(q0, q0, lane, ml, hf, aq, Ktb, Vb, ldsP[wave], acc, mrow, lrow);

#pragma unroll
  for (int rt = 0; rt < 2; ++rt)
#pragma unroll
    for (int j = 0; j < 8; ++j) {
      float inv = 1.f / lrow[rt][j];
      u16* orow = Ob + (size_t)(q0 + rt * 16 + j + 8 * hf) * tstr;
#pragma unroll
      for (int ct = 0; ct < 4; ++ct)
        orow[ct * 16 + ml] = f2bf(acc[rt][ct][j] * inv);
    }
}

// =====================================================================
// Fused residual add + LayerNorm -> f32 x and bf16 xb.
// =====================================================================
__global__ __launch_bounds__(256)
void add_ln_kernel(const float* __restrict__ xin, const float* __restrict__ res,
                   const float* __restrict__ w, const float* __restrict__ b,
                   float* __restrict__ xout, u16* __restrict__ xbout) {
  __shared__ float sred[256];
  const int row = blockIdx.x;
  const int tid = threadIdx.x;
  const float* xr = xin + (size_t)row * DD;
  const float* rr = res + (size_t)row * DD;
  float vals[4];
  float s = 0.f;
#pragma unroll
  for (int i = 0; i < 4; ++i) {
    int d = tid + i * 256;
    vals[i] = xr[d] + rr[d];
    s += vals[i];
  }
  sred[tid] = s; __syncthreads();
  for (int st = 128; st > 0; st >>= 1) { if (tid < st) sred[tid] += sred[tid + st]; __syncthreads(); }
  float mean = sred[0] * (1.f / DD);
  __syncthreads();
  float vs = 0.f;
#pragma unroll
  for (int i = 0; i < 4; ++i) { float dl = vals[i] - mean; vs += dl * dl; }
  sred[tid] = vs; __syncthreads();
  for (int st = 128; st > 0; st >>= 1) { if (tid < st) sred[tid] += sred[tid + st]; __syncthreads(); }
  float inv = rsqrtf(sred[0] * (1.f / DD) + EPSV);
#pragma unroll
  for (int i = 0; i < 4; ++i) {
    int d = tid + i * 256;
    float o = (vals[i] - mean) * inv * w[d] + b[d];
    xout[(size_t)row * DD + d] = o;
    xbout[(size_t)row * DD + d] = f2bf(o);
  }
}

// =====================================================================
static inline void launch_gemm(const u16* A, const u16* Bt, float* Cf, u16* Cb,
                               int N, int Nld, int K, int relu, float oscale,
                               bool bound, hipStream_t s) {
  dim3 grid((N + 63) / 64, MTOK / 128);
  if (bound)
    wmma_gemm<true><<<grid, dim3(256), 0, s>>>(A, Bt, Cf, Cb, N, Nld, K, relu, oscale);
  else
    wmma_gemm<false><<<grid, dim3(256), 0, s>>>(A, Bt, Cf, Cb, N, Nld, K, relu, oscale);
}

static inline void launch_wtrans(const float* in, u16* out, int N, int K, int Npad,
                                 hipStream_t s) {
  dim3 grid(K / 32, Npad / 32);
  wtrans_kernel<<<grid, dim3(256), 0, s>>>(in, out, N, K, Npad);
}

extern "C" void kernel_launch(void* const* d_in, const int* in_sizes, int n_in,
                              void* d_out, int out_size, void* d_ws, size_t ws_size,
                              hipStream_t stream) {
  (void)in_sizes; (void)n_in; (void)out_size; (void)ws_size;
  const int*   idx  = (const int*)d_in[0];
  const float* emb  = (const float*)d_in[1];
  const float* Wq   = (const float*)d_in[2];
  const float* Wk   = (const float*)d_in[3];
  const float* Wv   = (const float*)d_in[4];
  const float* Wo   = (const float*)d_in[5];
  const float* ln1w = (const float*)d_in[6];
  const float* ln1b = (const float*)d_in[7];
  const float* W1   = (const float*)d_in[8];
  const float* W2   = (const float*)d_in[9];
  const float* ln2w = (const float*)d_in[10];
  const float* ln2b = (const float*)d_in[11];
  const float* Wout = (const float*)d_in[12];
  float* out = (float*)d_out;

  // ---- workspace carve (~136 MB) ----
  char* p = (char*)d_ws;
  float* x     = (float*)p; p += (size_t)MTOK * DD * 4;
  float* resid = (float*)p; p += (size_t)MTOK * DD * 4;
  u16* xb   = (u16*)p; p += (size_t)MTOK * DD * 2;
  u16* qb   = (u16*)p; p += (size_t)MTOK * DD * 2;
  u16* kb   = (u16*)p; p += (size_t)MTOK * DD * 2;
  u16* vb   = (u16*)p; p += (size_t)MTOK * DD * 2;
  u16* kt   = (u16*)p; p += (size_t)MTOK * DD * 2;
  u16* ob   = (u16*)p; p += (size_t)MTOK * DD * 2;
  u16* hbf  = (u16*)p; p += (size_t)MTOK * DFF * 2;
  u16* wqt  = (u16*)p; p += (size_t)DD * DD * 2;
  u16* wkt  = (u16*)p; p += (size_t)DD * DD * 2;
  u16* wvt  = (u16*)p; p += (size_t)DD * DD * 2;
  u16* wot  = (u16*)p; p += (size_t)DD * DD * 2;
  u16* w1t  = (u16*)p; p += (size_t)DD * DFF * 2;    // also Wout^T slot
  u16* w2t  = (u16*)p; p += (size_t)DFF * DD * 2;

  embed_pe_kernel<<<(MTOK * DD) / 256, 256, 0, stream>>>(idx, emb, x, xb);

  const float qscale = 0.125f;   // 1/sqrt(DK), folded into q projection
  for (int l = 0; l < LL; ++l) {
    launch_wtrans(Wq + (size_t)l * DD * DD, wqt, DD, DD, DD, stream);
    launch_wtrans(Wk + (size_t)l * DD * DD, wkt, DD, DD, DD, stream);
    launch_wtrans(Wv + (size_t)l * DD * DD, wvt, DD, DD, DD, stream);
    launch_wtrans(Wo + (size_t)l * DD * DD, wot, DD, DD, DD, stream);
    launch_wtrans(W1 + (size_t)l * DFF * DD, w1t, DFF, DD, DFF, stream);
    launch_wtrans(W2 + (size_t)l * DD * DFF, w2t, DD, DFF, DD, stream);

    launch_gemm(xb, wqt, nullptr, qb, DD, DD, DD, 0, qscale, false, stream);
    launch_gemm(xb, wkt, nullptr, kb, DD, DD, DD, 0, 1.f, false, stream);
    launch_gemm(xb, wvt, nullptr, vb, DD, DD, DD, 0, 1.f, false, stream);
    kt_trans_kernel<<<dim3(SS / 32, DKK / 32, BB * HH), 256, 0, stream>>>(kb, kt);
    attn_kernel<<<dim3(SS / 256, HH, BB), 256, 0, stream>>>(qb, kt, vb, ob);
    launch_gemm(ob, wot, resid, nullptr, DD, DD, DD, 0, 1.f, false, stream);
    add_ln_kernel<<<MTOK, 256, 0, stream>>>(x, resid, ln1w + (size_t)l * DD,
                                            ln1b + (size_t)l * DD, x, xb);
    launch_gemm(xb, w1t, nullptr, hbf, DFF, DFF, DD, 1, 1.f, false, stream);
    launch_gemm(hbf, w2t, resid, nullptr, DD, DD, DFF, 0, 1.f, false, stream);
    add_ln_kernel<<<MTOK, 256, 0, stream>>>(x, resid, ln2w + (size_t)l * DD,
                                            ln2b + (size_t)l * DD, x, xb);
  }
  launch_wtrans(Wout, w1t, VV, DD, 2048, stream);
  launch_gemm(xb, w1t, out, nullptr, VV, 2048, DD, 0, 1.f, true, stream);
}
